// EGI_80719615361651
// MI455X (gfx1250) — matrix operand order repaired and checked by reading, not compiled, not run on hardware
//
#include <hip/hip_runtime.h>
#include <stdint.h>

// ---------------------------------------------------------------------------
// B=8, C=256, HW=4096, CE=32, CQ=16.
// out = sod + (wv @ (sod @ softmax(q^T k)^T) + bv) * sod
// (bias bv folds exactly: softmax rows sum to 1 -> attention applied to raw
//  sod, wv GEMM as epilogue).
// ---------------------------------------------------------------------------
#define HW 4096
#define NB 8
#define NC 256
#define CE 32

typedef __attribute__((ext_vector_type(16))) __bf16         v16bf;
typedef __attribute__((ext_vector_type(8)))  float          v8f;
typedef __attribute__((ext_vector_type(4)))  unsigned int   v4u;
typedef __attribute__((ext_vector_type(4)))  float          v4f;
typedef __attribute__((ext_vector_type(4)))  unsigned short v4s;

#if defined(__has_builtin)
#if __has_builtin(__builtin_amdgcn_cvt_pk_bf16_f32)
#define HAVE_CVT_PK_BF16 1
#endif
#endif

__device__ __forceinline__ unsigned short f2bf(float f) {
  unsigned int u = __float_as_uint(f);
  u += 0x7fffu + ((u >> 16) & 1u);          // round-to-nearest-even
  return (unsigned short)(u >> 16);
}

__device__ __forceinline__ unsigned int pk_bf16(float lo, float hi) {
#ifdef HAVE_CVT_PK_BF16
  typedef __attribute__((ext_vector_type(2))) __bf16 v2bf;
  union { v2bf v; unsigned int u; } t;
  t.v = __builtin_amdgcn_cvt_pk_bf16_f32(lo, hi);   // v_cvt_pk_bf16_f32
  return t.u;
#else
  return (unsigned int)f2bf(lo) | ((unsigned int)f2bf(hi) << 16);
#endif
}

union Frag16 {                              // 16 bf16 = 8 VGPRs per lane
  v4u          u4[2];
  unsigned int u[8];
  unsigned short s[16];
  v16bf        v;
};

// Async stage of one 64B V row chunk (32 bf16, j contiguous) into LDS.
// INST_OFFSET is added to BOTH the LDS and global address (ISA 08 §4.4).
__device__ __forceinline__ void stage_row64(unsigned lds_off,
                                            const unsigned short* g) {
  unsigned long long ga = (unsigned long long)(uintptr_t)g;
  asm volatile("global_load_async_to_lds_b128 %0, %1, off"
               :: "v"(lds_off), "v"(ga) : "memory");
  asm volatile("global_load_async_to_lds_b128 %0, %1, off offset:16"
               :: "v"(lds_off), "v"(ga) : "memory");
  asm volatile("global_load_async_to_lds_b128 %0, %1, off offset:32"
               :: "v"(lds_off), "v"(ga) : "memory");
  asm volatile("global_load_async_to_lds_b128 %0, %1, off offset:48"
               :: "v"(lds_off), "v"(ga) : "memory");
}

// ---------------------------------------------------------------------------
// Kernel 1: q/k 1x1 convs -> bf16, padded-K layout [B*HW, 32] (d=16..31 = 0).
// ---------------------------------------------------------------------------
__global__ void k_prep_qk(const float* __restrict__ edge,
                          const float* __restrict__ wq, const float* __restrict__ bq,
                          const float* __restrict__ wk, const float* __restrict__ bk,
                          unsigned short* __restrict__ qT,
                          unsigned short* __restrict__ kT) {
  int t = blockIdx.x * blockDim.x + threadIdx.x;   // [0, NB*HW)
  int b = t >> 12, i = t & (HW - 1);
  float e[CE];
  const float* ep = edge + ((size_t)b * CE) * HW + i;
#pragma unroll
  for (int c = 0; c < CE; ++c) e[c] = ep[(size_t)c * HW];
  unsigned short* qr = qT + ((size_t)t << 5);
  unsigned short* kr = kT + ((size_t)t << 5);
#pragma unroll
  for (int d = 0; d < 16; ++d) {
    float aq = bq[d], ak = bk[d];
#pragma unroll
    for (int c = 0; c < CE; ++c) {
      aq += wq[d * CE + c] * e[c];
      ak += wk[d * CE + c] * e[c];
    }
    qr[d] = f2bf(aq);  kr[d] = f2bf(ak);
    qr[16 + d] = 0;    kr[16 + d] = 0;               // K padding 16..31
  }
}

// ---------------------------------------------------------------------------
// Kernel 2: sod f32 [B,C,HW] -> bf16 same layout ("V" for attention).
// ---------------------------------------------------------------------------
__global__ void k_prep_sod(const float* __restrict__ s,
                           unsigned short* __restrict__ d) {
  size_t t = (size_t)blockIdx.x * blockDim.x + threadIdx.x;  // one per 4 elems
  v4f f = ((const v4f*)s)[t];
  v4s o;
#pragma unroll
  for (int r = 0; r < 4; ++r) o[r] = f2bf(f[r]);
  ((v4s*)d)[t] = o;
}

// ---------------------------------------------------------------------------
// Kernel 3: pre-swizzle wv (256x256) into WMMA B-fragment-major bf16 order.
// ---------------------------------------------------------------------------
__global__ void k_prep_wv(const float* __restrict__ wv,
                          unsigned short* __restrict__ wvf) {
  int f = blockIdx.x * blockDim.x + threadIdx.x;   // [0, 65536)
  int e    = f & 15;
  int lane = (f >> 4) & 31;
  int nb   = (f >> 9) & 15;
  int kb   = f >> 13;
  int o  = (nb << 4) + (lane & 15);
  int cp = (kb << 5) + ((e >> 3) << 4) + ((lane >> 4) << 3) + (e & 7);
  wvf[f] = f2bf(wv[o * NC + cp]);
}

// ---------------------------------------------------------------------------
// Kernel 4: flash attention.  Block = 8 waves = 8 i-tiles of one batch; each
// wave owns 16 query rows x all 256 channels (16 f32 accum fragments).
// Scores computed transposed (A = k tile [16j x 32d], B = q tile) so the
// score D-fragment IS the bf16 A-fragment element order of P -> no transpose.
// V tile (256c x 32j, 16KB) staged into double-buffered LDS once per block
// with global_load_async_to_lds_b128 (ASYNCcnt), shared by all 8 waves.
// Single barrier per iteration: each wave waits for its OWN async loads at
// the end of the body (a full iteration of overlap), so the top-of-loop
// barrier both publishes buffer p and WAR-protects buffer p^1.
// Accumulator rescale is vote-gated (running max rarely changes).
// ---------------------------------------------------------------------------
__global__ void __launch_bounds__(256) k_attn(
    const unsigned short* __restrict__ qT,    // [B*HW, 32] bf16 (padded d)
    const unsigned short* __restrict__ kT,    // [B*HW, 32] bf16 (padded d)
    const unsigned short* __restrict__ vbf,   // [B, C, HW] bf16
    unsigned short* __restrict__ attn)        // [B*HW, 256] bf16
{
  // 2 buffers x 256 rows x (64B data + 16B pad) -> conflict-free ds_load_b128
  __shared__ __align__(128) unsigned short smem[2][NC][40];

  const int tid  = threadIdx.x;
  const int lane = tid & 31;
  const int h    = lane >> 4;
  const int lm   = lane & 15;
  const int wid  = tid >> 5;
  const int b    = blockIdx.x >> 5;                       // batch (uniform)
  const int i0   = (((blockIdx.x & 31) << 3) + wid) << 4; // 16 query rows

  // --- async V staging: thread t owns channel row c = t -------------------
  const unsigned short* vrow = vbf + (size_t)(b * NC + tid) * HW;
  stage_row64((unsigned)(uintptr_t)&smem[0][tid][0], vrow);   // j0 = 0
  asm volatile("s_wait_asynccnt 0x0" ::: "memory");           // prologue pub.

  // B-fragment of Q (loop invariant): lane: N=i=i0+lm, K=d=8h+e (e<8), 16+8h+(e-8) (e>=8)
  Frag16 fq;
  {
    const unsigned short* p = qT + ((size_t)(b * HW + i0 + lm) << 5);
    fq.u4[0] = *(const v4u*)(p + (h << 3));
    fq.u4[1] = *(const v4u*)(p + 16 + (h << 3));
  }
  const unsigned short* kbase = kT + ((size_t)(b * HW) << 5);

  v8f acc[16];
#pragma unroll
  for (int ct = 0; ct < 16; ++ct) acc[ct] = (v8f){0,0,0,0,0,0,0,0};
  float m = -3.0e38f, l = 0.f;

  int p = 0;
  for (int j0 = 0; j0 < HW; j0 += 32) {
    // publishes buffer p (everyone waited on own loads last iteration) and
    // WAR-protects buffer p^1 (everyone finished reading it last iteration)
    __syncthreads();
    stage_row64((unsigned)(uintptr_t)&smem[p ^ 1][tid][0],
                vrow + ((j0 + 32) & (HW - 1)));   // stage next tile

    // prefetch K rows two j-blocks ahead (global_prefetch_b8)
    __builtin_prefetch(kbase + ((size_t)((j0 + 64 + lm) & (HW - 1)) << 5), 0, 3);

    // ---- QK: A-fragments of K (row j = j0(+16)+lm) -----------------------
    Frag16 fk0, fk1;
    {
      const unsigned short* r0 = kbase + ((size_t)(j0 + lm) << 5);
      fk0.u4[0] = *(const v4u*)(r0 + (h << 3));
      fk0.u4[1] = *(const v4u*)(r0 + 16 + (h << 3));
      const unsigned short* r1 = kbase + ((size_t)(j0 + 16 + lm) << 5);
      fk1.u4[0] = *(const v4u*)(r1 + (h << 3));
      fk1.u4[1] = *(const v4u*)(r1 + 16 + (h << 3));
    }
    v8f z = {0,0,0,0,0,0,0,0};
    v8f s0 = __builtin_amdgcn_wmma_f32_16x16x32_bf16(false, fk0.v, false, fq.v,
                                                     (short)0, z, false, false);
    v8f s1 = __builtin_amdgcn_wmma_f32_16x16x32_bf16(false, fk1.v, false, fq.v,
                                                     (short)0, z, false, false);

    // ---- online softmax for row i = lm ----------------------------------
    float bm = s0[0];
#pragma unroll
    for (int r = 0; r < 8; ++r) { bm = fmaxf(bm, s0[r]); bm = fmaxf(bm, s1[r]); }
    bm = fmaxf(bm, __shfl_xor(bm, 16));
    float mn = fmaxf(m, bm);

    if (__any(bm > m)) {          // scalar branch; EXEC stays all-ones
      float alpha = __expf(m - mn);
      l *= alpha;
      float al8[8];
#pragma unroll
      for (int r = 0; r < 8; ++r) al8[r] = __shfl(alpha, (h << 3) + r);
#pragma unroll
      for (int ct = 0; ct < 16; ++ct)
#pragma unroll
        for (int r = 0; r < 8; ++r) acc[ct][r] *= al8[r];
    }

    float p0a[8], p1a[8], ls = 0.f;
#pragma unroll
    for (int r = 0; r < 8; ++r) {
      p0a[r] = __expf(s0[r] - mn);
      p1a[r] = __expf(s1[r] - mn);
      ls += p0a[r] + p1a[r];
    }
    Frag16 fp;                    // P as bf16 A-fragment (element-order match)
#pragma unroll
    for (int g = 0; g < 4; ++g) {
      fp.u[g]     = pk_bf16(p0a[2 * g], p0a[2 * g + 1]);
      fp.u[4 + g] = pk_bf16(p1a[2 * g], p1a[2 * g + 1]);
    }
    ls += __shfl_xor(ls, 16);
    l += ls;
    m = mn;

    // ---- P x V from published LDS buffer p -------------------------------
#pragma unroll
    for (int ct = 0; ct < 16; ++ct) {
      Frag16 fv;   // B-frag from LDS: lane: N=c=16ct+lm, K=j=8h+e (e<8), 16+8h+(e-8) (e>=8)
      const unsigned short* lr = &smem[p][(ct << 4) + lm][0];
      fv.u4[0] = *(const v4u*)(lr + (h << 3));
      fv.u4[1] = *(const v4u*)(lr + 16 + (h << 3));
      acc[ct] = __builtin_amdgcn_wmma_f32_16x16x32_bf16(false, fp.v, false, fv.v,
                                                        (short)0, acc[ct],
                                                        false, false);
    }

    // own next-buffer loads complete (full body of overlap behind them)
    asm volatile("s_wait_asynccnt 0x0" ::: "memory");
    p ^= 1;
  }

  // normalize by row sums; store attn[b, i, c] bf16 (row-major, 256 c)
  float il[8];
#pragma unroll
  for (int r = 0; r < 8; ++r) il[r] = 1.f / __shfl(l, (h << 3) + r);

  unsigned short* arow = attn + ((size_t)(b * HW + i0) << 8) + lm;
#pragma unroll
  for (int ct = 0; ct < 16; ++ct) {
#pragma unroll
    for (int r = 0; r < 8; ++r) {
      int iloc = (h << 3) + r;
      arow[((size_t)iloc << 8) + (ct << 4)] = f2bf(acc[ct][r] * il[r]);
    }
  }
}

// ---------------------------------------------------------------------------
// Kernel 5: feat = wv @ attn + bv ; out = sod * (1 + feat).
// ---------------------------------------------------------------------------
__global__ void __launch_bounds__(256) k_out(
    const unsigned short* __restrict__ attn,  // [B*HW, 256] bf16
    const unsigned short* __restrict__ wvf,   // fragment-major bf16
    const float* __restrict__ bv,
    const float* __restrict__ sod,
    float* __restrict__ out)
{
  const int lane = threadIdx.x & 31;
  const int h = lane >> 4, lm = lane & 15;
  const int wg = (blockIdx.x << 3) + (threadIdx.x >> 5);   // [0, 32768)
  const int ob = wg & 15;
  const int it = (wg >> 4) & 255;
  const int b  = wg >> 12;
  const int i0 = it << 4;

  v8f acc = {0,0,0,0,0,0,0,0};
#pragma unroll
  for (int kb = 0; kb < 8; ++kb) {
    Frag16 fa, fw;
    const unsigned short* ar =
        attn + ((size_t)(b * HW + i0 + lm) << 8) + (kb << 5);
    fa.u4[0] = *(const v4u*)(ar + (h << 3));
    fa.u4[1] = *(const v4u*)(ar + 16 + (h << 3));
    const unsigned short* wp =
        wvf + (size_t)((kb << 4) + ob) * 512 + (lane << 4);
    fw.u4[0] = ((const v4u*)wp)[0];
    fw.u4[1] = ((const v4u*)wp)[1];
    acc = __builtin_amdgcn_wmma_f32_16x16x32_bf16(false, fa.v, false, fw.v,
                                                  (short)0, acc, false, false);
  }

  const int o = (ob << 4) + lm;
  const float bvo = bv[o];
  const size_t base = (size_t)(b * NC + o) * HW + i0 + (h << 3);
  v4f s0 = *(const v4f*)(sod + base);
  v4f s1 = *(const v4f*)(sod + base + 4);
  v4f o0, o1;
#pragma unroll
  for (int r = 0; r < 4; ++r) o0[r] = s0[r] + (acc[r] + bvo) * s0[r];
#pragma unroll
  for (int r = 0; r < 4; ++r) o1[r] = s1[r] + (acc[4 + r] + bvo) * s1[r];
  *(v4f*)(out + base)     = o0;
  *(v4f*)(out + base + 4) = o1;
}

// ---------------------------------------------------------------------------
extern "C" void kernel_launch(void* const* d_in, const int* in_sizes, int n_in,
                              void* d_out, int out_size, void* d_ws, size_t ws_size,
                              hipStream_t stream) {
  const float* sod  = (const float*)d_in[0];
  const float* edge = (const float*)d_in[1];
  const float* wq   = (const float*)d_in[2];
  const float* bq   = (const float*)d_in[3];
  const float* wk   = (const float*)d_in[4];
  const float* bk   = (const float*)d_in[5];
  const float* wv   = (const float*)d_in[6];
  const float* bv   = (const float*)d_in[7];

  unsigned short* qT   = (unsigned short*)d_ws;
  unsigned short* kT   = qT  + (size_t)NB * HW * 32;
  unsigned short* vbf  = kT  + (size_t)NB * HW * 32;
  unsigned short* wvf  = vbf + (size_t)NB * NC * HW;
  unsigned short* attn = wvf + (size_t)NC * NC;

  k_prep_qk <<<(NB * HW) / 256, 256, 0, stream>>>(edge, wq, bq, wk, bk, qT, kT);
  k_prep_sod<<<(NB * NC * HW / 4) / 256, 256, 0, stream>>>(sod, vbf);
  k_prep_wv <<<(NC * NC) / 256, 256, 0, stream>>>(wv, wvf);
  k_attn    <<<256, 256, 0, stream>>>(qT, kT, vbf, attn);
  k_out     <<<4096, 256, 0, stream>>>(attn, wvf, bv, sod, (float*)d_out);
}
